// DeepFCNet_14078902796795
// MI455X (gfx1250) — compile-verified, matrix operand order checked by reading, not run on hardware
//
#include <hip/hip_runtime.h>
#include <hip/hip_bf16.h>

typedef __attribute__((ext_vector_type(16))) _Float16 v16h;
typedef __attribute__((ext_vector_type(8)))  float    v8f;

#define BN_SCALE_C   0.9999950000374997f        // 1/sqrt(1+1e-5)
#define RRELU_SLOPE_C 0.22916666666666666f      // (1/8 + 1/3)/2
#define SIM_ROWS 271350                          // 30 * 9045
#define NPAIRS 9045

// CDNA5 async global->LDS path (ASYNCcnt), guarded so a missing builtin
// falls back to a synchronous cooperative fill.
#if defined(__has_builtin)
#if __has_builtin(__builtin_amdgcn_global_load_async_to_lds_b128) && \
    __has_builtin(__builtin_amdgcn_s_wait_asynccnt)
#define USE_ASYNC_LDS 1
#endif
#endif
#ifndef USE_ASYNC_LDS
#define USE_ASYNC_LDS 0
#endif

// Builtin signature (from hipcc diagnostic): pointee is GCC-style int4
// vector; AS1 (global) source, AS3 (LDS) dest.
typedef int gv4i __attribute__((vector_size(16)));
typedef __attribute__((address_space(1))) gv4i as1_gv4i;
typedef __attribute__((address_space(3))) gv4i as3_gv4i;

// ---------------------------------------------------------------------------
// WMMA helpers (CDNA5 f16 16x16x32, f32 accumulate)
// ---------------------------------------------------------------------------
__device__ __forceinline__ v8f wmma_f16(v16h a, v16h b, v8f c) {
  return __builtin_amdgcn_wmma_f32_16x16x32_f16(
      /*neg_a=*/false, a, /*neg_b=*/false, b,
      /*c_mod=*/(short)0, c, /*reuse_a=*/false, /*reuse_b=*/false);
}

// A tile 16(M) x 32(K) from f16 LDS, row-major, leading dim ld.
__device__ __forceinline__ v16h fragA_lds(const _Float16* t, int ld) {
  const int lane = threadIdx.x & 31;
  const int m  = lane & 15;
  const int ko = (lane >> 4) << 3;
  v16h a;
#pragma unroll
  for (int h = 0; h < 2; ++h)
#pragma unroll
    for (int v = 0; v < 4; ++v) {
      int k = h * 16 + ko + v * 2;
      a[h * 8 + v * 2 + 0] = t[m * ld + k];
      a[h * 8 + v * 2 + 1] = t[m * ld + k + 1];
    }
  return a;
}

// B tile 32(K) x 16(N) from f16 LDS, row-major (K rows), leading dim ld.
__device__ __forceinline__ v16h fragB_lds(const _Float16* t, int ld) {
  const int lane = threadIdx.x & 31;
  const int n  = lane & 15;
  const int ko = (lane >> 4) << 3;
  v16h b;
#pragma unroll
  for (int h = 0; h < 2; ++h)
#pragma unroll
    for (int v = 0; v < 4; ++v) {
      int k = h * 16 + ko + v * 2;
      b[h * 8 + v * 2 + 0] = t[k * ld + n];
      b[h * 8 + v * 2 + 1] = t[(k + 1) * ld + n];
    }
  return b;
}

// f32-LDS variants (convert at register-build time; staging holds raw bytes)
__device__ __forceinline__ v16h fragA_lds32(const float* t, int ld) {
  const int lane = threadIdx.x & 31;
  const int m  = lane & 15;
  const int ko = (lane >> 4) << 3;
  v16h a;
#pragma unroll
  for (int h = 0; h < 2; ++h)
#pragma unroll
    for (int v = 0; v < 4; ++v) {
      int k = h * 16 + ko + v * 2;
      a[h * 8 + v * 2 + 0] = (_Float16)t[m * ld + k];
      a[h * 8 + v * 2 + 1] = (_Float16)t[m * ld + k + 1];
    }
  return a;
}
__device__ __forceinline__ v16h fragB_lds32(const float* t, int ld) {
  const int lane = threadIdx.x & 31;
  const int n  = lane & 15;
  const int ko = (lane >> 4) << 3;
  v16h b;
#pragma unroll
  for (int h = 0; h < 2; ++h)
#pragma unroll
    for (int v = 0; v < 4; ++v) {
      int k = h * 16 + ko + v * 2;
      b[h * 8 + v * 2 + 0] = (_Float16)t[k * ld + n];
      b[h * 8 + v * 2 + 1] = (_Float16)t[(k + 1) * ld + n];
    }
  return b;
}

// ---------------------------------------------------------------------------
// Kernel 1: build pair index tables (inverse of np.triu_indices(135, k=1))
// ---------------------------------------------------------------------------
__global__ void pair_kernel(int* __restrict__ pi, int* __restrict__ pj) {
  int p = blockIdx.x * blockDim.x + threadIdx.x;
  if (p >= NPAIRS) return;
  int i = 0, rem = p;
  while (rem >= 134 - i) { rem -= (134 - i); ++i; }
  pi[p] = i;
  pj[p] = i + 1 + rem;
}

// ---------------------------------------------------------------------------
// Kernel 2: fused feature extractor, one workgroup per (subject,node) sample.
// ---------------------------------------------------------------------------
#define FE_OFF_X   0u
#define FE_OFF_P1  1536u
#define FE_OFF_B2  7488u
#define FE_OFF_A2  32064u
#define FE_OFF_S2  40256u
#define FE_OFF_P2  87360u
#define FE_OFF_B3  93248u
#define FE_OFF_A3  111680u
#define FE_OFF_S3  130112u
#define FE_OFF_P3  164672u
#define FE_OFF_PART 173312u
#define FE_SMEM_BYTES 173824u
#define FE_THREADS 128

__global__ void __launch_bounds__(FE_THREADS) fe_kernel(
    const float* __restrict__ x,
    const float* __restrict__ w1, const float* __restrict__ b1,
    const float* __restrict__ g1, const float* __restrict__ bb1,
    const float* __restrict__ w2, const float* __restrict__ b2,
    const float* __restrict__ g2, const float* __restrict__ bb2,
    const float* __restrict__ w3, const float* __restrict__ b3,
    const float* __restrict__ g3, const float* __restrict__ bb3,
    const float* __restrict__ fcw, const float* __restrict__ fcb,
    float* __restrict__ feats)
{
  extern __shared__ char smem[];
  float*    xrow = (float*)   (smem + FE_OFF_X);    // 375
  _Float16* p1   = (_Float16*)(smem + FE_OFF_P1);   // 16 x 186
  _Float16* B2   = (_Float16*)(smem + FE_OFF_B2);   // 64 x 192
  _Float16* A2   = (_Float16*)(smem + FE_OFF_A2);   // 64 x 64
  float*    s2   = (float*)   (smem + FE_OFF_S2);   // 64 x 184
  _Float16* p2   = (_Float16*)(smem + FE_OFF_P2);   // 32 x 92
  _Float16* B3   = (_Float16*)(smem + FE_OFF_B3);   // 96 x 96
  _Float16* A3   = (_Float16*)(smem + FE_OFF_A3);   // 96 x 96
  float*    s3   = (float*)   (smem + FE_OFF_S3);   // 96 x 90
  float*    p3   = (float*)   (smem + FE_OFF_P3);   // 48 x 45
  float*    part = (float*)   (smem + FE_OFF_PART); // 128

  const int s    = blockIdx.x;
  const int tid  = threadIdx.x;
  const int wave = tid >> 5;
  const int lane = tid & 31;

  for (int i = tid; i < 375; i += FE_THREADS) xrow[i] = x[s * 375 + i];
  __syncthreads();

  // conv1 + BN + leaky + 2x2 pool over (ch, t) -> p1 (16 x 186)
  for (int idx = tid; idx < 16 * 186; idx += FE_THREADS) {
    int c = idx / 186, t = idx % 186;
    float mx = -3.4e38f;
#pragma unroll
    for (int dc = 0; dc < 2; ++dc) {
      int cc = 2 * c + dc;
      float sc = g1[cc] * BN_SCALE_C, sb = bb1[cc], cb = b1[cc];
      float wk0 = w1[cc * 3 + 0], wk1 = w1[cc * 3 + 1], wk2 = w1[cc * 3 + 2];
#pragma unroll
      for (int dt = 0; dt < 2; ++dt) {
        int tt = 2 * t + dt;
        float v = cb + wk0 * xrow[tt] + wk1 * xrow[tt + 1] + wk2 * xrow[tt + 2];
        v = v * sc + sb;
        v = v > 0.f ? v : 0.01f * v;
        mx = fmaxf(mx, v);
      }
    }
    p1[c * 186 + t] = (_Float16)mx;
  }
  __syncthreads();

  // im2col B2 (K=48 pad 64) x (N=184 pad 192); A2 = W2 (64 x 48 pad 64)
  for (int idx = tid; idx < 64 * 192; idx += FE_THREADS) {
    int k = idx / 192, n = idx % 192;
    _Float16 v = (_Float16)0.f;
    if (k < 48 && n < 184) { int ch = k / 3, tap = k % 3; v = p1[ch * 186 + n + tap]; }
    B2[idx] = v;
  }
  for (int idx = tid; idx < 64 * 64; idx += FE_THREADS) {
    int k = idx & 63;
    A2[idx] = (k < 48) ? (_Float16)w2[(idx >> 6) * 48 + k] : (_Float16)0.f;
  }
  __syncthreads();

  // conv2 GEMM: 4 m-tiles x 12 n-tiles, 2 k-steps
  for (int tile = wave; tile < 48; tile += 4) {
    int mt = tile / 12, nt = tile % 12;
    v8f acc = {};
#pragma unroll
    for (int ks = 0; ks < 2; ++ks) {
      v16h a = fragA_lds(A2 + mt * 16 * 64 + ks * 32, 64);
      v16h b = fragB_lds(B2 + ks * 32 * 192 + nt * 16, 192);
      acc = wmma_f16(a, b, acc);
    }
    int n = nt * 16 + (lane & 15);
    int mbase = mt * 16 + ((lane >> 4) << 3);
    if (n < 184) {
#pragma unroll
      for (int r = 0; r < 8; ++r) {
        int m = mbase + r;
        float v = acc[r] + b2[m];
        v = v * (g2[m] * BN_SCALE_C) + bb2[m];
        v = v > 0.f ? v : 0.01f * v;
        s2[m * 184 + n] = v;
      }
    }
  }
  __syncthreads();

  // pool2 -> p2 (32 x 92)
  for (int idx = tid; idx < 32 * 92; idx += FE_THREADS) {
    int c = idx / 92, t = idx % 92;
    float v = fmaxf(fmaxf(s2[(2 * c) * 184 + 2 * t], s2[(2 * c) * 184 + 2 * t + 1]),
                    fmaxf(s2[(2 * c + 1) * 184 + 2 * t], s2[(2 * c + 1) * 184 + 2 * t + 1]));
    p2[idx] = (_Float16)v;
  }
  __syncthreads();

  // im2col B3 (K=96) x (N=90 pad 96); A3 = W3 (96 x 96)
  for (int idx = tid; idx < 96 * 96; idx += FE_THREADS) {
    int k = idx / 96, n = idx % 96;
    _Float16 v = (_Float16)0.f;
    if (n < 90) { int ch = k / 3, tap = k % 3; v = p2[ch * 92 + n + tap]; }
    B3[idx] = v;
    A3[idx] = (_Float16)w3[idx];   // (96,32,3) flat == m*96 + ch*3+tap
  }
  __syncthreads();

  // conv3 GEMM: 6 m-tiles x 6 n-tiles, 3 k-steps
  for (int tile = wave; tile < 36; tile += 4) {
    int mt = tile / 6, nt = tile % 6;
    v8f acc = {};
#pragma unroll
    for (int ks = 0; ks < 3; ++ks) {
      v16h a = fragA_lds(A3 + mt * 16 * 96 + ks * 32, 96);
      v16h b = fragB_lds(B3 + ks * 32 * 96 + nt * 16, 96);
      acc = wmma_f16(a, b, acc);
    }
    int n = nt * 16 + (lane & 15);
    int mbase = mt * 16 + ((lane >> 4) << 3);
    if (n < 90) {
#pragma unroll
      for (int r = 0; r < 8; ++r) {
        int m = mbase + r;
        float v = acc[r] + b3[m];
        v = v * (g3[m] * BN_SCALE_C) + bb3[m];
        v = v > 0.f ? v : 0.01f * v;
        s3[m * 90 + n] = v;
      }
    }
  }
  __syncthreads();

  // pool3 -> p3 (48 x 45)
  for (int idx = tid; idx < 48 * 45; idx += FE_THREADS) {
    int c = idx / 45, t = idx % 45;
    float v = fmaxf(fmaxf(s3[(2 * c) * 90 + 2 * t], s3[(2 * c) * 90 + 2 * t + 1]),
                    fmaxf(s3[(2 * c + 1) * 90 + 2 * t], s3[(2 * c + 1) * 90 + 2 * t + 1]));
    p3[idx] = v;
  }
  __syncthreads();

  // FC 2160 -> 32
  {
    int c = tid & 31, seg = tid >> 5;
    float acc = 0.f;
    for (int i = seg * 540; i < seg * 540 + 540; ++i)
      acc += p3[i] * fcw[i * 32 + c];
    part[seg * 32 + c] = acc;
  }
  __syncthreads();
  if (tid < 32)
    feats[s * 32 + tid] = part[tid] + part[32 + tid] + part[64 + tid] + part[96 + tid] + fcb[tid];
}

// ---------------------------------------------------------------------------
// Kernel 3: all-pairs concat + 4-layer similarity MLP, fused per 16-row tile.
// ---------------------------------------------------------------------------
__global__ void __launch_bounds__(256) sim_kernel(
    const float* __restrict__ feats,
    const int* __restrict__ pit, const int* __restrict__ pjt,
    const float* __restrict__ w1, const float* __restrict__ b1,
    const float* __restrict__ w2, const float* __restrict__ b2,
    const float* __restrict__ w3, const float* __restrict__ b3,
    const float* __restrict__ w4, const float* __restrict__ b4,
    float* __restrict__ subjects)
{
  __shared__ _Float16 W1s[64 * 32];
  __shared__ _Float16 W2s[32 * 16];
  __shared__ _Float16 W3s[32 * 16];
  __shared__ float sb1[32], sb2[16], sb3[8], sw4[8], sb4[1];
  __shared__ _Float16 hbuf[8][16 * 32];

  const int tid = threadIdx.x;
  for (int i = tid; i < 64 * 32; i += 256) W1s[i] = (_Float16)w1[i];
  for (int i = tid; i < 32 * 16; i += 256) W2s[i] = (_Float16)w2[i];
  for (int i = tid; i < 32 * 16; i += 256) {
    int k = i >> 4, n = i & 15;
    W3s[i] = (k < 16 && n < 8) ? (_Float16)w3[k * 8 + n] : (_Float16)0.f;
  }
  if (tid < 32) sb1[tid] = b1[tid];
  if (tid < 16) sb2[tid] = b2[tid];
  if (tid < 8)  { sb3[tid] = b3[tid]; sw4[tid] = w4[tid]; }
  if (tid == 0) sb4[0] = b4[0];
  __syncthreads();

  const int wave = tid >> 5;
  const int lane = tid & 31;
  const int rowbase = blockIdx.x * 128 + wave * 16;
  _Float16* hw = hbuf[wave];

  const int m  = lane & 15;
  const int ko = (lane >> 4) << 3;
  const int row = rowbase + m;
  const float* f0 = nullptr;
  const float* f1 = nullptr;
  if (row < SIM_ROWS) {
    int b = row / NPAIRS, p = row - b * NPAIRS;
    f0 = feats + (b * 135 + pit[p]) * 32;
    f1 = feats + (b * 135 + pjt[p]) * 32;
  }
  v16h a1a, a1b;
#pragma unroll
  for (int h = 0; h < 2; ++h)
#pragma unroll
    for (int v = 0; v < 4; ++v)
#pragma unroll
      for (int q = 0; q < 2; ++q) {
        int kl = h * 16 + ko + v * 2 + q;
        int idx = h * 8 + v * 2 + q;
        a1a[idx] = (_Float16)(f0 ? f0[kl] : 0.f);
        a1b[idx] = (_Float16)(f1 ? f1[kl] : 0.f);
      }

  v8f acc0 = {}, acc1 = {};
  acc0 = wmma_f16(a1a, fragB_lds(W1s + 0 * 32 * 32 + 0, 32), acc0);
  acc0 = wmma_f16(a1b, fragB_lds(W1s + 1 * 32 * 32 + 0, 32), acc0);
  acc1 = wmma_f16(a1a, fragB_lds(W1s + 0 * 32 * 32 + 16, 32), acc1);
  acc1 = wmma_f16(a1b, fragB_lds(W1s + 1 * 32 * 32 + 16, 32), acc1);

  const int hi8 = (lane >> 4) << 3;
  const int nn  = lane & 15;
#pragma unroll
  for (int r = 0; r < 8; ++r) {
    int mm = r + hi8;
    float v0 = acc0[r] + sb1[nn];
    float v1 = acc1[r] + sb1[16 + nn];
    hw[mm * 32 + nn]      = (_Float16)(v0 > 0.f ? v0 : 0.f);
    hw[mm * 32 + 16 + nn] = (_Float16)(v1 > 0.f ? v1 : 0.f);
  }

  {
    v16h a2 = fragA_lds(hw, 32);
    v8f z = {};
    v8f acc2 = wmma_f16(a2, fragB_lds(W2s, 16), z);
#pragma unroll
    for (int r = 0; r < 8; ++r) {
      int mm = r + hi8;
      float v = acc2[r] + sb2[nn];
      hw[mm * 32 + nn]      = (_Float16)(v > 0.f ? v : 0.f);
      hw[mm * 32 + 16 + nn] = (_Float16)0.f;
    }
  }

  {
    v16h a3 = fragA_lds(hw, 32);
    v8f z = {};
    v8f acc3 = wmma_f16(a3, fragB_lds(W3s, 16), z);
#pragma unroll
    for (int r = 0; r < 8; ++r) {
      int mm = r + hi8;
      if (nn < 8) {
        float v = acc3[r] + sb3[nn];
        hw[mm * 32 + nn] = (_Float16)(v > 0.f ? v : 0.f);
      }
    }
  }

  if (lane < 16) {
    int rrow = rowbase + lane;
    if (rrow < SIM_ROWS) {
      float sacc = sb4[0];
#pragma unroll
      for (int k = 0; k < 8; ++k) sacc += (float)hw[lane * 32 + k] * sw4[k];
      subjects[rrow] = tanhf(sacc);
    }
  }
}

// ---------------------------------------------------------------------------
// Kernels 4-6: WMMA GEMM, M padded 30->32, LDS-staged A/B stripes.
// Block = 8 waves: mt = wave&1 (2 m-tiles), ntl = wave>>1 (4 n-tiles);
// grid.x = N/64. Per k-step the block stages A(32x32) and B(32x64) f32
// stripes into LDS — via GLOBAL_LOAD_ASYNC_TO_LDS_B128 (ASYNCcnt) when the
// stripe is fully in-bounds, synchronous zero-padded fill on the K tail.
// ---------------------------------------------------------------------------
__global__ void __launch_bounds__(256) cls_gemm_kernel(
    const float* __restrict__ A, const float* __restrict__ W,
    const float* __restrict__ bias, float* __restrict__ out,
    int Mvalid, int K, int N, float neg_slope)
{
  __shared__ float As[32 * 32];   // 32 m x 32 k
  __shared__ float Bs[32 * 64];   // 32 k x 64 n
  const int tid  = threadIdx.x;
  const int wave = tid >> 5;
  const int lane = tid & 31;
  const int mt   = wave & 1;
  const int ntl  = wave >> 1;          // 0..3
  const int nbase = blockIdx.x * 64;
  const int ksteps = (K + 31) >> 5;
  v8f acc = {};

  for (int ks = 0; ks < ksteps; ++ks) {
    const int kb = ks << 5;
    __syncthreads();                   // protect previous step's frag reads
#if USE_ASYNC_LDS
    if (kb + 32 <= K) {
      // B stripe: 32 rows x 256 B, 2 x b128 per lane (ASYNCcnt tracked).
#pragma unroll
      for (int j = 0; j < 2; ++j) {
        int q = tid + j * 256;
        int kr = q >> 4, c4 = (q & 15) << 2;
        const float* g = W + (size_t)(kb + kr) * N + nbase + c4;
        float* l = Bs + kr * 64 + c4;
        __builtin_amdgcn_global_load_async_to_lds_b128(
            (as1_gv4i*)(void*)g, (as3_gv4i*)(void*)l, 0, 0);
      }
      // A stripe: 32 rows x 128 B, 1 x b128 per lane. OOB M rows (30,31)
      // only feed discarded D rows.
      {
        int mr = tid >> 3, c4 = (tid & 7) << 2;
        const float* g = A + (size_t)mr * K + kb + c4;
        float* l = As + mr * 32 + c4;
        __builtin_amdgcn_global_load_async_to_lds_b128(
            (as1_gv4i*)(void*)g, (as3_gv4i*)(void*)l, 0, 0);
      }
      __builtin_prefetch(W + (size_t)(kb + 32) * N + nbase, 0, 1);
      __builtin_amdgcn_s_wait_asynccnt(0);
    } else
#endif
    {
      for (int idx = tid; idx < 32 * 64; idx += 256) {
        int kr = idx >> 6, nc = idx & 63;
        int kg = kb + kr;
        Bs[idx] = (kg < K) ? W[(size_t)kg * N + nbase + nc] : 0.f;
      }
      for (int idx = tid; idx < 32 * 32; idx += 256) {
        int mr = idx >> 5, kc = idx & 31;
        int kg = kb + kc;
        As[idx] = (mr < Mvalid && kg < K) ? A[(size_t)mr * K + kg] : 0.f;
      }
    }
    __syncthreads();
    v16h af = fragA_lds32(As + mt * 16 * 32, 32);
    v16h bf = fragB_lds32(Bs + ntl * 16, 64);
    acc = wmma_f16(af, bf, acc);
  }

  const int hi8 = (lane >> 4) << 3;
  const int nn = nbase + ntl * 16 + (lane & 15);
#pragma unroll
  for (int r = 0; r < 8; ++r) {
    int mm = mt * 16 + r + hi8;
    if (mm < Mvalid) {
      float v = acc[r] + bias[nn];
      v = v >= 0.f ? v : neg_slope * v;
      out[(size_t)mm * N + nn] = v;
    }
  }
}

// ---------------------------------------------------------------------------
// Kernel 7: final 64 -> 3 + log_softmax
// ---------------------------------------------------------------------------
__global__ void cls4_kernel(const float* __restrict__ h,
                            const float* __restrict__ w,
                            const float* __restrict__ b,
                            float* __restrict__ out)
{
  int t = threadIdx.x;
  if (t >= 30) return;
  float v[3];
#pragma unroll
  for (int o = 0; o < 3; ++o) {
    float sacc = b[o];
    for (int k = 0; k < 64; ++k) sacc += h[t * 64 + k] * w[k * 3 + o];
    v[o] = sacc;
  }
  float mx = fmaxf(v[0], fmaxf(v[1], v[2]));
  float se = expf(v[0] - mx) + expf(v[1] - mx) + expf(v[2] - mx);
  float lse = mx + logf(se);
#pragma unroll
  for (int o = 0; o < 3; ++o) out[t * 3 + o] = v[o] - lse;
}

// ---------------------------------------------------------------------------
extern "C" void kernel_launch(void* const* d_in, const int* in_sizes, int n_in,
                              void* d_out, int out_size, void* d_ws, size_t ws_size,
                              hipStream_t stream) {
  const float* x      = (const float*)d_in[0];
  const float* c1w    = (const float*)d_in[1];
  const float* c1b    = (const float*)d_in[2];
  const float* g1     = (const float*)d_in[3];
  const float* bb1    = (const float*)d_in[4];
  const float* c2w    = (const float*)d_in[5];
  const float* c2b    = (const float*)d_in[6];
  const float* g2     = (const float*)d_in[7];
  const float* bb2    = (const float*)d_in[8];
  const float* c3w    = (const float*)d_in[9];
  const float* c3b    = (const float*)d_in[10];
  const float* g3     = (const float*)d_in[11];
  const float* bb3    = (const float*)d_in[12];
  const float* fcw    = (const float*)d_in[13];
  const float* fcb    = (const float*)d_in[14];
  const float* sw1    = (const float*)d_in[15];
  const float* sbb1   = (const float*)d_in[16];
  const float* sw2    = (const float*)d_in[17];
  const float* sbb2   = (const float*)d_in[18];
  const float* sw3    = (const float*)d_in[19];
  const float* sbb3   = (const float*)d_in[20];
  const float* sw4    = (const float*)d_in[21];
  const float* sbb4   = (const float*)d_in[22];
  const float* clw1   = (const float*)d_in[23];
  const float* clb1   = (const float*)d_in[24];
  const float* clw2   = (const float*)d_in[25];
  const float* clb2   = (const float*)d_in[26];
  const float* clw3   = (const float*)d_in[27];
  const float* clb3   = (const float*)d_in[28];
  const float* clw4   = (const float*)d_in[29];
  const float* clb4   = (const float*)d_in[30];
  float* out = (float*)d_out;

  char* ws = (char*)d_ws;
  int*   pi       = (int*)  (ws + 0);
  int*   pj       = (int*)  (ws + 36192);
  float* feats    = (float*)(ws + 72384);                       // 4050 x 32
  float* subjects = (float*)(ws + 590784);                      // 30 x 9045
  float* hc1      = (float*)(ws + 1676192);                     // 30 x 1024
  float* hc2      = (float*)(ws + 1799072);                     // 30 x 256
  float* hc3      = (float*)(ws + 1829792);                     // 30 x 64

  pair_kernel<<<(NPAIRS + 255) / 256, 256, 0, stream>>>(pi, pj);

  fe_kernel<<<4050, FE_THREADS, FE_SMEM_BYTES, stream>>>(
      x, c1w, c1b, g1, bb1, c2w, c2b, g2, bb2, c3w, c3b, g3, bb3,
      fcw, fcb, feats);

  sim_kernel<<<(SIM_ROWS + 127) / 128, 256, 0, stream>>>(
      feats, pi, pj, sw1, sbb1, sw2, sbb2, sw3, sbb3, sw4, sbb4, subjects);

  cls_gemm_kernel<<<16, 256, 0, stream>>>(subjects, clw1, clb1, hc1,
                                          30, 9045, 1024, RRELU_SLOPE_C);
  cls_gemm_kernel<<<4, 256, 0, stream>>>(hc1, clw2, clb2, hc2,
                                         30, 1024, 256, 0.0f);
  cls_gemm_kernel<<<1, 256, 0, stream>>>(hc2, clw3, clb3, hc3,
                                         30, 256, 64, 0.0f);

  cls4_kernel<<<1, 32, 0, stream>>>(hc3, clw4, clb4, out);

  (void)in_sizes; (void)n_in; (void)out_size; (void)ws_size;
}
